// ScaledDotProductAttention_29824252903756
// MI455X (gfx1250) — matrix operand order repaired and checked by reading, not compiled, not run on hardware
//
#include <hip/hip_runtime.h>
#include <hip/hip_bf16.h>

// Problem constants (match reference)
#define BB 2
#define HH 16
#define SS 2048
#define DD 64
#define QT 16            // query rows per workgroup
#define SROW (SS + 4)    // padded LDS row stride (floats) to dodge bank conflicts
#define THR2 (11.3f * 11.3f)
#define NEGBIG (-1e9f)

typedef __attribute__((ext_vector_type(16))) __bf16 v16bf;
typedef __attribute__((ext_vector_type(8)))  float  v8f;

__device__ __forceinline__ __bf16 bhi(float x) { return (__bf16)x; }
__device__ __forceinline__ __bf16 blo(float x) {
  __bf16 h = (__bf16)x;
  return (__bf16)(x - (float)h);
}

__global__ __launch_bounds__(128) void fused_dist_attn_kernel(
    const float* __restrict__ Q, const float* __restrict__ K,
    const float* __restrict__ V, const unsigned char* __restrict__ Mask,
    float* __restrict__ Out, float* __restrict__ Attn)
{
  __shared__ float sQ[QT * 68];       // padded Q tile (16x64)
  __shared__ float sS[QT * SROW];     // full score/prob rows, 16 x 2048 f32
  __shared__ float sK2[SS];           // ||k||^2 per key
  __shared__ float sV[2][32 * DD];    // double-buffered async-staged V chunks
  __shared__ float sQ2[QT];           // ||q||^2 per query row
  __shared__ float sRed[QT * 8];      // partial reductions
  __shared__ float sMax[QT];
  __shared__ float sInv[QT];          // 1 / row sum

  const int tid  = threadIdx.x;
  const int lane = tid & 31;
  const int wave = tid >> 5;          // 0..3
  const int nlo  = lane & 15;         // N / M-low index within wave
  const int hk   = lane >> 4;         // lane-half selector

  const int bh = blockIdx.x >> 7;     // S/QT = 128 q-tiles per head
  const int qt = blockIdx.x & 127;
  const int q0 = qt * QT;

  const size_t headBase = (size_t)bh * SS * DD;
  const float* Qh = Q + headBase + (size_t)q0 * DD;
  const float* Kh = K + headBase;
  const float* Vh = V + headBase;
  float* AttnH = Attn + (size_t)bh * SS * SS + (size_t)q0 * SS;

  // ---- stage Q tile ----
  for (int e = tid; e < QT * DD; e += 128) {
    int m = e >> 6, d = e & 63;
    sQ[m * 68 + d] = Qh[(size_t)m * DD + d];
  }
  __syncthreads();

  // ---- q^2 (16 rows) ----
  if (tid < QT) {
    float s = 0.f;
#pragma unroll
    for (int d = 0; d < DD; ++d) { float x = sQ[tid * 68 + d]; s += x * x; }
    sQ2[tid] = s;
  }
  // ---- k^2 for all keys (reads hit L2) ----
  for (int c = tid; c < SS; c += 128) {
    const float4* kp = (const float4*)(Kh + (size_t)c * DD);
    float s = 0.f;
#pragma unroll
    for (int j = 0; j < DD / 4; ++j) {
      float4 f = kp[j];
      s += f.x * f.x + f.y * f.y + f.z * f.z + f.w * f.w;
    }
    sK2[c] = s;
  }
  __syncthreads();

  // ---- persistent A (Q) fragments: 2 K-chunks x (hi,lo) bf16 split ----
  // A k order used here (matches B): M = lane&15; element i -> K = i + 16*(lane>>4)
  v16bf a_hi[2], a_lo[2];
#pragma unroll
  for (int c = 0; c < 2; ++c) {
    int base = nlo * 68 + c * 32 + hk * 16;
#pragma unroll
    for (int i = 0; i < 16; ++i) {
      float x = sQ[base + i];
      a_hi[c][i] = bhi(x);
      a_lo[c][i] = blo(x);
    }
  }

  // =========== phase 1: scores = QK^T, cdist mask, scale ===========
  for (int t = 0; t < 32; ++t) {
    const int col0 = (wave * 32 + t) * 16;
    if (t + 1 < 32)
      __builtin_prefetch(Kh + (size_t)(col0 + 16 + nlo) * DD, 0, 1);

    v8f acc = {};
#pragma unroll
    for (int c = 0; c < 2; ++c) {
      // B k order used here (matches A): N = lane&15; element i -> K = i + 16*(lane>>4)
      const float4* kp =
          (const float4*)(Kh + (size_t)(col0 + nlo) * DD + c * 32 + hk * 16);
      float kv[16];
#pragma unroll
      for (int j = 0; j < 4; ++j) {
        float4 f = kp[j];
        kv[4 * j + 0] = f.x; kv[4 * j + 1] = f.y;
        kv[4 * j + 2] = f.z; kv[4 * j + 3] = f.w;
      }
      v16bf b_hi, b_lo;
#pragma unroll
      for (int i = 0; i < 16; ++i) { b_hi[i] = bhi(kv[i]); b_lo[i] = blo(kv[i]); }

      // bf16x3 emulation of fp32 dot: Ah*Bh + Ah*Bl + Al*Bh
      acc = __builtin_amdgcn_wmma_f32_16x16x32_bf16(false, a_hi[c], false, b_hi,
                                                    (short)0, acc, false, false);
      acc = __builtin_amdgcn_wmma_f32_16x16x32_bf16(false, a_hi[c], false, b_lo,
                                                    (short)0, acc, false, false);
      acc = __builtin_amdgcn_wmma_f32_16x16x32_bf16(false, a_lo[c], false, b_hi,
                                                    (short)0, acc, false, false);
    }

    // C/D layout: col N = lane&15, row M = i + 8*(lane>>4)
    const int n = col0 + nlo;
    const float k2 = sK2[n];
#pragma unroll
    for (int i = 0; i < 8; ++i) {
      int m = i + 8 * hk;
      float qk = acc[i];
      float d2 = sQ2[m] + k2 - 2.f * qk;                 // cdist^2 (>=thr same as sqrt>=11.3)
      bool msk = (Mask[(size_t)(q0 + m) * SS + n] != 0) | (d2 >= THR2);
      sS[m * SROW + n] = msk ? NEGBIG : qk * 0.125f;     // 1/sqrt(64)
    }
  }
  __syncthreads();

  // =========== phase 2: row softmax in LDS ===========
  const int r = tid >> 3, sub = tid & 7;   // 8 threads per row
  {
    float mx = -3.4e38f;
    for (int c = sub * 256; c < sub * 256 + 256; ++c)
      mx = fmaxf(mx, sS[r * SROW + c]);
    sRed[r * 8 + sub] = mx;
  }
  __syncthreads();
  if (sub == 0) {
    float mx = sRed[r * 8];
#pragma unroll
    for (int j = 1; j < 8; ++j) mx = fmaxf(mx, sRed[r * 8 + j]);
    sMax[r] = mx;
  }
  __syncthreads();
  {
    const float rm = sMax[r];
    float sm = 0.f;
    for (int c = sub * 256; c < sub * 256 + 256; ++c) {
      float p = __expf(sS[r * SROW + c] - rm);
      sS[r * SROW + c] = p;                 // keep un-normalized p for PV
      sm += p;
    }
    sRed[r * 8 + sub] = sm;
  }
  __syncthreads();
  if (sub == 0) {
    float sm = sRed[r * 8];
#pragma unroll
    for (int j = 1; j < 8; ++j) sm += sRed[r * 8 + j];
    sInv[r] = 1.0f / sm;
  }
  __syncthreads();

  // ---- kick off async staging of V chunk 0 while we drain the weights ----
  // CDNA5 async copy: GLOBAL_LOAD_ASYNC_TO_LDS_B128, tracked by ASYNCcnt.
  // LDS dest address = low 32 bits of the generic shared pointer.
  {
    unsigned ldsbase = (unsigned)(unsigned long long)(void*)&sV[0][0];
    const float* src = Vh;  // chunk 0
#pragma unroll
    for (int g0 = 0; g0 < 4; ++g0) {
      int g = tid + g0 * 128;                // 512 x 16B groups = 32x64 f32
      unsigned lds  = ldsbase + (unsigned)g * 16u;
      unsigned voff = (unsigned)g * 16u;
      asm volatile("global_load_async_to_lds_b128 %0, %1, %2"
                   :: "v"(lds), "v"(voff), "s"(src) : "memory");
    }
  }

  // ---- normalized weights out (dominant HBM traffic, coalesced) ----
  for (int r2 = 0; r2 < QT; ++r2) {
    const float inv = sInv[r2];
    for (int c = tid; c < SS; c += 128)
      AttnH[(size_t)r2 * SS + c] = sS[r2 * SROW + c] * inv;
  }

  // =========== phase 3: O = P V (each wave owns 16 output columns) ===========
  v8f o = {};
  const int ng = wave * 16 + nlo;          // global output column
  for (int kc = 0; kc < SS / 32; ++kc) {
    const int buf = kc & 1;
    // my async copies done; then barrier so every wave's copies are visible
    asm volatile("s_wait_asynccnt 0x0" ::: "memory");
    __syncthreads();
    // prefetch next V chunk into the other buffer (overlaps the WMMAs below)
    if (kc + 1 < SS / 32) {
      unsigned ldsbase = (unsigned)(unsigned long long)(void*)&sV[buf ^ 1][0];
      const float* src = Vh + (size_t)(kc + 1) * 32 * DD;
#pragma unroll
      for (int g0 = 0; g0 < 4; ++g0) {
        int g = tid + g0 * 128;
        unsigned lds  = ldsbase + (unsigned)g * 16u;
        unsigned voff = (unsigned)g * 16u;
        asm volatile("global_load_async_to_lds_b128 %0, %1, %2"
                     :: "v"(lds), "v"(voff), "s"(src) : "memory");
      }
    }

    v16bf p_hi, p_lo;
    {
      int base = nlo * SROW + kc * 32 + hk * 16;
#pragma unroll
      for (int i = 0; i < 16; ++i) {
        float x = sS[base + i];
        p_hi[i] = bhi(x);
        p_lo[i] = blo(x);
      }
    }
    v16bf v_hi, v_lo;
    {
#pragma unroll
      for (int i = 0; i < 16; ++i) {
        float x = sV[buf][(hk * 16 + i) * DD + ng];
        v_hi[i] = bhi(x);
        v_lo[i] = blo(x);
      }
    }
    o = __builtin_amdgcn_wmma_f32_16x16x32_bf16(false, p_hi, false, v_hi,
                                                (short)0, o, false, false);
    o = __builtin_amdgcn_wmma_f32_16x16x32_bf16(false, p_hi, false, v_lo,
                                                (short)0, o, false, false);
    o = __builtin_amdgcn_wmma_f32_16x16x32_bf16(false, p_lo, false, v_hi,
                                                (short)0, o, false, false);
  }
#pragma unroll
  for (int i = 0; i < 8; ++i) {
    int m = i + 8 * hk;
    Out[headBase + (size_t)(q0 + m) * DD + ng] = o[i] * sInv[m];
  }
}

extern "C" void kernel_launch(void* const* d_in, const int* in_sizes, int n_in,
                              void* d_out, int out_size, void* d_ws, size_t ws_size,
                              hipStream_t stream) {
  (void)in_sizes; (void)n_in; (void)d_ws; (void)ws_size; (void)out_size;
  const float* q = (const float*)d_in[0];
  const float* k = (const float*)d_in[1];
  const float* v = (const float*)d_in[2];
  const unsigned char* mask = (const unsigned char*)d_in[3];
  float* out  = (float*)d_out;                              // [B,H,S,D]
  float* attn = out + (size_t)BB * HH * SS * DD;            // [B,H,S,S]

  dim3 grid(BB * HH * (SS / QT));   // 4096 workgroups
  dim3 block(128);                  // 4 wave32
  fused_dist_attn_kernel<<<grid, block, 0, stream>>>(q, k, v, mask, out, attn);
}
